// CircuitGNN_Polarity_40484361732243
// MI455X (gfx1250) — compile-verified, matrix operand order checked by reading, not run on hardware
//
#include <hip/hip_runtime.h>

typedef __attribute__((ext_vector_type(16))) _Float16 v16h;
typedef __attribute__((ext_vector_type(8)))  _Float16 v8h;
typedef __attribute__((ext_vector_type(8)))  float    v8f;

#define NN      50000
#define FIN     17
#define HID     64
#define NHEADS  2
#define NLAYERS 12
#define NEDGE   400000
#define ETOT    (NEDGE + NN)   // edges + self loops

__device__ __forceinline__ float elu1(float v) { return v > 0.f ? v : __expf(v) - 1.f; }

// ---------------------------------------------------------------- utilities
__global__ void k_zero(float* __restrict__ p, int n) {
  int i = blockIdx.x * blockDim.x + threadIdx.x;
  if (i < n) p[i] = 0.f;
}

// ---------------------------------------------------------------- input projection  x[N,17] @ W[17,64] + b
__global__ void k_in_proj(const float* __restrict__ x, const float* __restrict__ W,
                          const float* __restrict__ b, float* __restrict__ out) {
  int n = blockIdx.x * 4 + threadIdx.y;
  int c = threadIdx.x;
  if (n >= NN) return;
  const float* xr = x + (size_t)n * FIN;
  float acc = b[c];
#pragma unroll
  for (int k = 0; k < FIN; ++k) acc += xr[k] * W[k * HID + c];
  out[(size_t)n * HID + c] = acc;
}

// ---------------------------------------------------------------- batchnorm (training-mode batch stats)
__global__ void k_bn_stats(const float* __restrict__ t, float* __restrict__ gsum,
                           float* __restrict__ gsq) {
  __shared__ float ssum[4][HID];
  __shared__ float ssq[4][HID];
  int c = threadIdx.x, ty = threadIdx.y;
  float s = 0.f, q = 0.f;
  for (int r = blockIdx.x * 4 + ty; r < NN; r += gridDim.x * 4) {
    float v = t[(size_t)r * HID + c];
    s += v; q += v * v;
  }
  ssum[ty][c] = s; ssq[ty][c] = q;
  __syncthreads();
  if (ty == 0) {
    s = ssum[0][c] + ssum[1][c] + ssum[2][c] + ssum[3][c];
    q = ssq[0][c] + ssq[1][c] + ssq[2][c] + ssq[3][c];
    atomicAdd(&gsum[c], s);
    atomicAdd(&gsq[c], q);
  }
}

__global__ void k_bn_finalize(const float* __restrict__ gsum, const float* __restrict__ gsq,
                              const float* __restrict__ gamma, const float* __restrict__ beta,
                              float* __restrict__ scale, float* __restrict__ shift) {
  int c = threadIdx.x;
  float mean = gsum[c] * (1.f / NN);
  float var  = gsq[c]  * (1.f / NN) - mean * mean;
  float sc   = gamma[c] * rsqrtf(var + 1e-5f);
  scale[c] = sc;
  shift[c] = beta[c] - mean * sc;
}

// BN + ELU (+ residual), fused with f16 repack for the next WMMA GEMM
__global__ void k_bn_apply(const float* __restrict__ t, const float* __restrict__ scale,
                           const float* __restrict__ shift, float* __restrict__ h,
                           _Float16* __restrict__ h16, int residual) {
  int i = blockIdx.x * blockDim.x + threadIdx.x;
  if (i >= NN * HID) return;
  int c = i & (HID - 1);
  float v = t[i] * scale[c] + shift[c];
  v = elu1(v);
  if (residual) v += h[i];
  h[i]  = v;
  h16[i] = (_Float16)v;
}

// ---------------------------------------------------------------- pack W[64,128] f32 -> Wt[128,64] f16 (both Wl, Wr)
__global__ void k_pack_w(const float* __restrict__ Wl, const float* __restrict__ Wr,
                         _Float16* __restrict__ wt) {
  int i = blockIdx.x * blockDim.x + threadIdx.x;
  if (i >= 2 * 128 * HID) return;
  int m  = i >> 13;       // which matrix
  int j  = i & 8191;
  int nn = j >> 6;        // output column 0..127
  int k  = j & 63;        // k 0..63
  const float* W = m ? Wr : Wl;
  wt[i] = (_Float16)W[k * 128 + nn];
}

// ---------------------------------------------------------------- C[N,128] = A[N,64] @ B[64,128] via v_wmma_f32_16x16x32_f16
// A16 row-major [N,64] f16; Bt16 = B^T row-major [128,64] f16.
// One wave computes one 16x16 C tile; 8 waves/block cover all 128 output cols.
__global__ void k_gemm_wmma(const _Float16* __restrict__ A16, const _Float16* __restrict__ Bt16,
                            float* __restrict__ C) {
  int lane = threadIdx.x & 31;
  int wave = threadIdx.x >> 5;      // 0..7 -> column tile
  int row0 = blockIdx.x * 16;
  int col0 = wave * 16;
  int r  = lane & 15;
  int hi = lane >> 4;               // lane half selects K sub-chunks per ISA layout
  const _Float16* arow = A16 + (size_t)(row0 + r) * HID;
  const _Float16* brow = Bt16 + (size_t)(col0 + r) * HID;
  v8f acc = {};
  union V { v16h v; v8h h[2]; };
#pragma unroll
  for (int kb = 0; kb < 2; ++kb) {  // K = 64 = 2 x 32
    int k0 = kb * 32 + hi * 8;
    V a, b;
    a.h[0] = *(const v8h*)(arow + k0);
    a.h[1] = *(const v8h*)(arow + k0 + 16);
    b.h[0] = *(const v8h*)(brow + k0);
    b.h[1] = *(const v8h*)(brow + k0 + 16);
    acc = __builtin_amdgcn_wmma_f32_16x16x32_f16(false, a.v, false, b.v,
                                                 (short)0, acc, false, false);
  }
#pragma unroll
  for (int i = 0; i < 8; ++i)
    C[(size_t)(row0 + i + 8 * hi) * 128 + col0 + r] = acc[i];
}

// ---------------------------------------------------------------- attention scores + segment max
// one thread per (edge, head); self loops are implicit for e >= NEDGE
__global__ void k_edge_scores(const int* __restrict__ ei, const float* __restrict__ xl,
                              const float* __restrict__ xr, const float* __restrict__ att,
                              float* __restrict__ ebuf, unsigned int* __restrict__ emax) {
  int idx = blockIdx.x * blockDim.x + threadIdx.x;
  if (idx >= ETOT * NHEADS) return;
  int e = idx >> 1, hd = idx & 1;
  int s, d;
  if (e < NEDGE) { s = ei[e]; d = ei[NEDGE + e]; } else { s = d = e - NEDGE; }
  const float4* pl = (const float4*)(xl + (size_t)s * 128 + hd * 64);
  const float4* pr = (const float4*)(xr + (size_t)d * 128 + hd * 64);
  const float4* pa = (const float4*)(att + hd * 64);
  float acc = 0.f;
#pragma unroll
  for (int i = 0; i < 16; ++i) {
    float4 a = pl[i], b = pr[i], t = pa[i];
    float v0 = a.x + b.x; v0 = v0 > 0.f ? v0 : 0.2f * v0;
    float v1 = a.y + b.y; v1 = v1 > 0.f ? v1 : 0.2f * v1;
    float v2 = a.z + b.z; v2 = v2 > 0.f ? v2 : 0.2f * v2;
    float v3 = a.w + b.w; v3 = v3 > 0.f ? v3 : 0.2f * v3;
    acc += v0 * t.x + v1 * t.y + v2 * t.z + v3 * t.w;
  }
  ebuf[idx] = acc;
  // order-preserving uint encoding for float atomicMax
  unsigned int bits = __float_as_uint(acc);
  unsigned int key  = (bits & 0x80000000u) ? ~bits : (bits | 0x80000000u);
  atomicMax(emax + (size_t)d * NHEADS + hd, key);
}

// exp(e - max) and segment-sum denominator
__global__ void k_edge_softmax(const int* __restrict__ ei, float* __restrict__ ebuf,
                               const unsigned int* __restrict__ emax, float* __restrict__ den) {
  int idx = blockIdx.x * blockDim.x + threadIdx.x;
  if (idx >= ETOT * NHEADS) return;
  int e = idx >> 1, hd = idx & 1;
  int d = (e < NEDGE) ? ei[NEDGE + e] : e - NEDGE;
  unsigned int u  = emax[(size_t)d * NHEADS + hd];
  unsigned int mb = (u & 0x80000000u) ? (u ^ 0x80000000u) : ~u;
  float m = __uint_as_float(mb);
  if (!__builtin_isfinite(m)) m = 0.f;   // empty-segment guard (matches reference)
  float ex = __expf(ebuf[idx] - m);
  ebuf[idx] = ex;
  atomicAdd(den + (size_t)d * NHEADS + hd, ex);
}

// weighted scatter-aggregation: agg[dst] += alpha * xl[src]; one wave per edge,
// each lane handles 4 contiguous channels (128 ch across 2 heads)
__global__ void k_aggregate(const int* __restrict__ ei, const float* __restrict__ xl,
                            const float* __restrict__ ebuf, const float* __restrict__ den,
                            float* __restrict__ agg) {
  int gid  = blockIdx.x * blockDim.x + threadIdx.x;
  int e    = gid >> 5;
  int lane = gid & 31;
  if (e >= ETOT) return;
  int s, d;
  if (e < NEDGE) { s = ei[e]; d = ei[NEDGE + e]; } else { s = d = e - NEDGE; }
  int hd  = lane >> 4;          // lanes 0-15: head 0, lanes 16-31: head 1
  int off = lane * 4;
  float alpha = ebuf[(size_t)e * NHEADS + hd] /
                (den[(size_t)d * NHEADS + hd] + 1e-16f);
  float4 v = *(const float4*)(xl + (size_t)s * 128 + off);
  float* o = agg + (size_t)d * 128 + off;
  atomicAdd(o + 0, alpha * v.x);
  atomicAdd(o + 1, alpha * v.y);
  atomicAdd(o + 2, alpha * v.z);
  atomicAdd(o + 3, alpha * v.w);
}

// mean over heads + conv bias
__global__ void k_head_mean(const float* __restrict__ agg, const float* __restrict__ bias,
                            float* __restrict__ tmp) {
  int i = blockIdx.x * blockDim.x + threadIdx.x;
  if (i >= NN * HID) return;
  int n = i >> 6, c = i & 63;
  tmp[i] = 0.5f * (agg[(size_t)n * 128 + c] + agg[(size_t)n * 128 + 64 + c]) + bias[c];
}

// ---------------------------------------------------------------- output MLP head + sigmoid
__global__ void k_out_head(const float* __restrict__ h, const float* __restrict__ W1,
                           const float* __restrict__ b1, const float* __restrict__ W2,
                           const float* __restrict__ b2, float* __restrict__ out) {
  int n = blockIdx.x * blockDim.x + threadIdx.x;
  if (n >= NN) return;
  const float* hr = h + (size_t)n * HID;
  float hv[HID];
#pragma unroll
  for (int k = 0; k < HID; ++k) hv[k] = hr[k];
  float acc2 = b2[0];
#pragma unroll 4
  for (int j = 0; j < 32; ++j) {
    float a = b1[j];
#pragma unroll
    for (int k = 0; k < HID; ++k) a += hv[k] * W1[k * 32 + j];
    a = elu1(a);
    acc2 += a * W2[j];
  }
  out[n] = 1.f / (1.f + __expf(-acc2));
}

// ---------------------------------------------------------------- driver
extern "C" void kernel_launch(void* const* d_in, const int* in_sizes, int n_in,
                              void* d_out, int out_size, void* d_ws, size_t ws_size,
                              hipStream_t stream) {
  (void)in_sizes; (void)n_in; (void)out_size; (void)ws_size;

  const float* x     = (const float*)d_in[0];
  const int*   ei    = (const int*)d_in[1];
  const float* in_W  = (const float*)d_in[2];
  const float* in_b  = (const float*)d_in[3];
  const float* in_g  = (const float*)d_in[4];
  const float* in_be = (const float*)d_in[5];
  const float* Wl    = (const float*)d_in[6];
  const float* Wr    = (const float*)d_in[7];
  const float* att   = (const float*)d_in[8];
  const float* cbias = (const float*)d_in[9];
  const float* bng   = (const float*)d_in[10];
  const float* bnb   = (const float*)d_in[11];
  const float* oW1   = (const float*)d_in[12];
  const float* ob1   = (const float*)d_in[13];
  const float* oW2   = (const float*)d_in[14];
  const float* ob2   = (const float*)d_in[15];

  // workspace layout (bytes), 256B-aligned sections; agg aliases xr
  char* ws = (char*)d_ws;
  float*        h    = (float*)(ws + 0);            // N*64 f32      12.8 MB
  float*        tmp  = (float*)(ws + 12800000);     // N*64 f32      12.8 MB
  _Float16*     h16  = (_Float16*)(ws + 25600000);  // N*64 f16       6.4 MB
  float*        xl   = (float*)(ws + 32000000);     // N*128 f32     25.6 MB
  float*        agg  = (float*)(ws + 57600000);     // N*128 f32     25.6 MB  (xr, then agg)
  float*        ebuf = (float*)(ws + 83200000);     // ETOT*2 f32     3.6 MB
  unsigned int* emax = (unsigned int*)(ws + 86800128); // N*2 u32     0.4 MB
  float*        den  = (float*)(ws + 87200128);     // N*2 f32        0.4 MB (contiguous after emax)
  _Float16*     w16  = (_Float16*)(ws + 87600128);  // 2*128*64 f16  32 KB
  float*        gsum = (float*)(ws + 87632896);     // 64
  float*        gsq  = gsum + 64;
  float*        scl  = gsum + 128;
  float*        shf  = gsum + 192;

  const int TPB = 256;
  dim3 b64x4(HID, 4);

  auto run_bn = [&](const float* g, const float* be, int residual) {
    k_zero<<<1, 128, 0, stream>>>(gsum, 128);
    k_bn_stats<<<128, b64x4, 0, stream>>>(tmp, gsum, gsq);
    k_bn_finalize<<<1, HID, 0, stream>>>(gsum, gsq, g, be, scl, shf);
    k_bn_apply<<<(NN * HID + TPB - 1) / TPB, TPB, 0, stream>>>(tmp, scl, shf, h, h16, residual);
  };

  // input projection + BN + ELU
  k_in_proj<<<(NN + 3) / 4, b64x4, 0, stream>>>(x, in_W, in_b, tmp);
  run_bn(in_g, in_be, 0);

  for (int i = 0; i < NLAYERS; ++i) {
    k_pack_w<<<(2 * 128 * HID + TPB - 1) / TPB, TPB, 0, stream>>>(
        Wl + (size_t)i * HID * 128, Wr + (size_t)i * HID * 128, w16);
    k_gemm_wmma<<<NN / 16, TPB, 0, stream>>>(h16, w16, xl);          // xl = h @ Wl
    k_gemm_wmma<<<NN / 16, TPB, 0, stream>>>(h16, w16 + 8192, agg);  // xr = h @ Wr
    // zero emax + den together (contiguous 200000 words)
    k_zero<<<(4 * NN + TPB - 1) / TPB, TPB, 0, stream>>>((float*)emax, 4 * NN);
    k_edge_scores<<<(ETOT * NHEADS + TPB - 1) / TPB, TPB, 0, stream>>>(
        ei, xl, agg, att + (size_t)i * NHEADS * HID, ebuf, emax);
    k_edge_softmax<<<(ETOT * NHEADS + TPB - 1) / TPB, TPB, 0, stream>>>(ei, ebuf, emax, den);
    k_zero<<<(NN * 128 + TPB - 1) / TPB, TPB, 0, stream>>>(agg, NN * 128);
    k_aggregate<<<(ETOT * 32 + TPB - 1) / TPB, TPB, 0, stream>>>(ei, xl, ebuf, den, agg);
    k_head_mean<<<(NN * HID + TPB - 1) / TPB, TPB, 0, stream>>>(agg, cbias + (size_t)i * HID, tmp);
    run_bn(bng + (size_t)i * HID, bnb + (size_t)i * HID, 1);
  }

  k_out_head<<<(NN + 63) / 64, 64, 0, stream>>>(h, oW1, ob1, oW2, ob2, (float*)d_out);
}
